// Encoder_1400159339188
// MI455X (gfx1250) — matrix operand order repaired and checked by reading, not compiled, not run on hardware
//
#include <hip/hip_runtime.h>
#include <hip/hip_bf16.h>

// ---------------- problem constants (fixed by the reference) ----------------
constexpr int FEAT_DIM = 256;
constexpr int CLIPS    = 8;
constexpr int DIM      = 128;
constexpr int K_NEIGH  = 32;
constexpr int NROWS    = 2048 * 10;       // B*S = 20480 flat nodes
constexpr int NCOLS    = CLIPS * DIM;     // 1024 output columns (c*128 + k)

// ---------------- CDNA5 WMMA types ----------------
typedef __attribute__((ext_vector_type(16))) __bf16 v16bf;
typedef __attribute__((ext_vector_type(8)))  float  v8f;
typedef __attribute__((ext_vector_type(4)))  int    i4;

struct U4  { unsigned int x, y, z, w; };         // 16-byte chunk
union  Frag { U4 u[2]; v16bf v; };               // 32 bytes = one 16-bit WMMA operand
struct US4 { unsigned short a, b, c, d; };       // 8-byte bf16x4 store

__device__ __forceinline__ unsigned short f2bf(float f) {
    __bf16 h = (__bf16)f;
    return __builtin_bit_cast(unsigned short, h);
}

// ---------------- async global->LDS copy (CDNA5, ASYNCcnt-tracked) ----------------
typedef __attribute__((address_space(1))) i4   gbl_i4;   // int4 in global memory
typedef __attribute__((address_space(3))) i4   lds_i4;   // int4 in LDS
typedef __attribute__((address_space(3))) void lds_void;

__device__ __forceinline__ void async_cp16(void* lds, const void* g) {
#if defined(__HIP_DEVICE_COMPILE__)
#if __has_builtin(__builtin_amdgcn_global_load_async_to_lds_b128)
    __builtin_amdgcn_global_load_async_to_lds_b128(
        (gbl_i4*)g, (lds_i4*)lds, /*imm offset*/0, /*cpol*/0);
#else
    unsigned           loff = (unsigned)(unsigned long long)(lds_void*)lds;
    unsigned long long ga   = (unsigned long long)g;
    asm volatile("global_load_async_to_lds_b128 %0, %1, off"
                 :: "v"(loff), "v"(ga) : "memory");
#endif
#endif
}

__device__ __forceinline__ void wait_async0() {
#if defined(__HIP_DEVICE_COMPILE__)
#if __has_builtin(__builtin_amdgcn_s_wait_asynccnt)
    __builtin_amdgcn_s_wait_asynccnt(0);
#else
    asm volatile("s_wait_asynccnt 0x0" ::: "memory");
#endif
#endif
}

// ---------------- kernel 1: W f32 [8,128,256] -> bf16 [1024,256] ----------------
__global__ void __launch_bounds__(256) wcvt_kernel(const float* __restrict__ w,
                                                   unsigned short* __restrict__ wbf) {
    int i = (blockIdx.x * 256 + threadIdx.x) * 4;
    const float* p = w + i;
    US4 o;
    o.a = f2bf(p[0]); o.b = f2bf(p[1]); o.c = f2bf(p[2]); o.d = f2bf(p[3]);
    *(US4*)(wbf + i) = o;
}

// ---------------- kernel 2: gather + mean -> bf16 feat [20480,256] ----------------
__global__ void __launch_bounds__(256) agg_kernel(const int* __restrict__ nodes,
                                                  const int* __restrict__ adj,
                                                  const float* __restrict__ features,
                                                  unsigned short* __restrict__ featbf) {
    const int lane = threadIdx.x & 31;
    const int wave = (blockIdx.x * 256 + threadIdx.x) >> 5;   // 0..20479 exact
    const int node = nodes[wave];
    const int myNeigh = adj[node * K_NEIGH + lane];           // one neighbor per lane

    float a0 = 0.f, a1 = 0.f, a2 = 0.f, a3 = 0.f;
    float b0 = 0.f, b1 = 0.f, b2 = 0.f, b3 = 0.f;

#pragma unroll 4
    for (int j = 0; j < K_NEIGH; ++j) {
        const int nb = __shfl(myNeigh, j, 32);                // wave32 broadcast
        const float* row = features + (size_t)nb * FEAT_DIM;
        const float4 lo = *(const float4*)(row + lane * 4);
        const float4 hi = *(const float4*)(row + 128 + lane * 4);
        a0 += lo.x; a1 += lo.y; a2 += lo.z; a3 += lo.w;
        b0 += hi.x; b1 += hi.y; b2 += hi.z; b3 += hi.w;
    }
    const float s = 1.0f / (float)K_NEIGH;
    unsigned short* dst = featbf + (size_t)wave * FEAT_DIM;
    US4 lo, hi;
    lo.a = f2bf(a0 * s); lo.b = f2bf(a1 * s); lo.c = f2bf(a2 * s); lo.d = f2bf(a3 * s);
    hi.a = f2bf(b0 * s); hi.b = f2bf(b1 * s); hi.c = f2bf(b2 * s); hi.d = f2bf(b3 * s);
    *(US4*)(dst + lane * 4)       = lo;
    *(US4*)(dst + 128 + lane * 4) = hi;
}

// ---------------- kernel 3: WMMA GEMM + ReLU, async-staged, A double-buffered ----
// out[n, j] = relu( sum_d featbf[n,d] * wbf[j,d] ),  n<20480, j<1024, d<256
constexpr int BM      = 128;              // block rows
constexpr int BN      = 64;               // block cols
constexpr int BK      = 32;               // K slice
constexpr int KSLICES = FEAT_DIM / BK;    // 8
constexpr int LDR     = 80;               // padded LDS row stride (64 data + 16)
constexpr int A_BUF   = BM * LDR;         // 10240 bytes per A buffer
constexpr int B_SLICE = BN * LDR;         // 5120 bytes per B K-slice
constexpr int B_OFF   = 2 * A_BUF;        // B region after double-buffered A
constexpr int SMEM_SZ = B_OFF + KSLICES * B_SLICE;   // 61440 bytes

__global__ void __launch_bounds__(256) gemm_kernel(const unsigned short* __restrict__ featbf,
                                                   const unsigned short* __restrict__ wbf,
                                                   float* __restrict__ out) {
    __shared__ __align__(16) unsigned char smem[SMEM_SZ];

    const int t     = threadIdx.x;
    const int lane  = t & 31;
    const int wid   = t >> 5;           // 0..7
    const int waveM = wid >> 1;         // 0..3  -> 32-row band
    const int waveN = wid & 1;          // 0..1  -> 32-col band
    const int rowBase = blockIdx.x * BM;
    const int colBase = blockIdx.y * BN;

    const int m = lane & 15;            // index within 16-tile
    const int h = lane >> 4;            // half-wave select
    const int seg  = t & 3;             // 16B segment within a 64B K-row

    // ---- issue all B slices (64 W-rows x full K) : 8 x 16B per thread ----
    {
        const int row = t >> 2;         // 0..63
        const unsigned short* g = wbf + (size_t)(colBase + row) * FEAT_DIM + seg * 8;
        unsigned char* l = smem + B_OFF + row * LDR + seg * 16;
#pragma unroll
        for (int s = 0; s < KSLICES; ++s)
            async_cp16(l + s * B_SLICE, g + s * BK);
    }
    // ---- issue A slice 0 into buffer 0 : 2 x 16B per thread, unconditional ----
    {
        const int r0 = t >> 2, r1 = (t + 256) >> 2;   // rows 0..63 and 64..127
        async_cp16(smem + r0 * LDR + seg * 16,
                   featbf + (size_t)(rowBase + r0) * FEAT_DIM + seg * 8);
        async_cp16(smem + r1 * LDR + seg * 16,
                   featbf + (size_t)(rowBase + r1) * FEAT_DIM + seg * 8);
    }
    wait_async0();
    __syncthreads();                    // A slice 0 + all B ready

    v8f acc[2][2] = {};                 // 2x2 tiles of 16x16 f32

    for (int ks = 0; ks < KSLICES; ++ks) {
        const int cur = ks & 1;
        // ---- overlap: issue next A slice into the other buffer ----
        if (ks + 1 < KSLICES) {
            const int nxt = cur ^ 1;
            const int kOff = (ks + 1) * BK;
            const int r0 = t >> 2, r1 = (t + 256) >> 2;
            async_cp16(smem + nxt * A_BUF + r0 * LDR + seg * 16,
                       featbf + (size_t)(rowBase + r0) * FEAT_DIM + kOff + seg * 8);
            async_cp16(smem + nxt * A_BUF + r1 * LDR + seg * 16,
                       featbf + (size_t)(rowBase + r1) * FEAT_DIM + kOff + seg * 8);
        }

        // ---- fragments (ISA VGPR layouts) ----
        // A lane(m,h): two 16B chunks at byte offsets h*16 and h*16+32 of the row
        v16bf aF[2], bF[2];
#pragma unroll
        for (int mi = 0; mi < 2; ++mi) {
            const unsigned char* p =
                smem + cur * A_BUF + (waveM * 32 + mi * 16 + m) * LDR + h * 16;
            Frag fr;
            fr.u[0] = *(const U4*)p;
            fr.u[1] = *(const U4*)(p + 32);
            aF[mi] = fr.v;
        }
        // B lane(n,h): two 16B chunks at byte offsets h*16 and h*16+32 of W-row n
#pragma unroll
        for (int ni = 0; ni < 2; ++ni) {
            const unsigned char* p =
                smem + B_OFF + ks * B_SLICE + (waveN * 32 + ni * 16 + m) * LDR + h * 16;
            Frag fr;
            fr.u[0] = *(const U4*)p;
            fr.u[1] = *(const U4*)(p + 32);
            bF[ni] = fr.v;
        }

#pragma unroll
        for (int mi = 0; mi < 2; ++mi)
#pragma unroll
            for (int ni = 0; ni < 2; ++ni)
                acc[mi][ni] = __builtin_amdgcn_wmma_f32_16x16x32_bf16(
                    false, aF[mi], false, bF[ni],
                    (short)0, acc[mi][ni], false, false);

        // ---- wait for next slice + make buffer swap safe ----
        if (ks + 1 < KSLICES) {
            wait_async0();
            __syncthreads();
        }
    }

    // ---- epilogue: ReLU + coalesced f32 stores ----
    // C/D layout: VGPR r, lanes 0-15 -> M=r, lanes 16-31 -> M=8+r; N = lane&15
#pragma unroll
    for (int mi = 0; mi < 2; ++mi) {
        const int rbase = rowBase + waveM * 32 + mi * 16 + h * 8;
#pragma unroll
        for (int ni = 0; ni < 2; ++ni) {
            const int col = colBase + waveN * 32 + ni * 16 + m;
#pragma unroll
            for (int r = 0; r < 8; ++r) {
                float v = acc[mi][ni][r];
                out[(size_t)(rbase + r) * NCOLS + col] = v > 0.f ? v : 0.f;
            }
        }
    }
}

// ---------------- host-side launch ----------------
extern "C" void kernel_launch(void* const* d_in, const int* in_sizes, int n_in,
                              void* d_out, int out_size, void* d_ws, size_t ws_size,
                              hipStream_t stream) {
    const int*   nodes        = (const int*)d_in[0];    // [2048,10]
    const int*   adj          = (const int*)d_in[1];    // [50000,32]
    const float* features     = (const float*)d_in[2];  // [50000,256]
    const float* local_weight = (const float*)d_in[3];  // [8,128,256]
    float*       out          = (float*)d_out;          // [20480,1024]

    // workspace: [0, 512KB) bf16 weights; then 10.5MB bf16 aggregated feats
    unsigned short* wbf    = (unsigned short*)d_ws;
    unsigned short* featbf = (unsigned short*)((char*)d_ws + (size_t)NCOLS * FEAT_DIM * 2);

    wcvt_kernel<<<(NCOLS * FEAT_DIM) / (256 * 4), 256, 0, stream>>>(local_weight, wbf);
    agg_kernel<<<NROWS / 8, 256, 0, stream>>>(nodes, adj, features, featbf);
    dim3 grid(NROWS / BM, NCOLS / BN);                  // 160 x 16
    gemm_kernel<<<grid, 256, 0, stream>>>(featbf, wbf, out);
}
